// LSTMEncoderDecoder_15461882266218
// MI455X (gfx1250) — matrix operand order repaired and checked by reading, not compiled, not run on hardware
//
#include <hip/hip_runtime.h>

// ---------------- problem constants ----------------
#define B_    1024
#define S_    256
#define IN_   8
#define H_    512
#define F_    96
#define G4_   2048          // 4*H
#define KAUG  544           // H + 32 (x lives in cols 512..519, rest zero pad)
#define KT_   17            // KAUG / 32 K-tiles per WMMA chain

typedef __attribute__((ext_vector_type(16))) __bf16 v16bf;
typedef __attribute__((ext_vector_type(8)))  __bf16 v8bf;
typedef __attribute__((ext_vector_type(8)))  float  v8f;

// ---- fast activations: v_rcp_f32 / v_tanh_f32 instead of IEEE div chains ----
__device__ __forceinline__ float rcp_(float x) {
#if __has_builtin(__builtin_amdgcn_rcpf)
    return __builtin_amdgcn_rcpf(x);
#else
    return 1.f / x;
#endif
}
__device__ __forceinline__ float sig_(float x) {
    return rcp_(1.f + __expf(-x));
}
__device__ __forceinline__ float tanh_(float x) {
#if __has_builtin(__builtin_amdgcn_tanhf)
    return __builtin_amdgcn_tanhf(x);
#else
    float e = __expf(2.f * x);
    return (e - 1.f) * rcp_(e + 1.f);
#endif
}

// ---------------- prep: build augmented bf16 weights + fused biases ----------------
// W_aug[n][k] : k<512 -> Whh[n][k]; 512<=k<512+IN -> Wih[n][k-512]; else 0
__global__ void pack_weights(const float* __restrict__ encWih, const float* __restrict__ encWhh,
                             const float* __restrict__ encbih, const float* __restrict__ encbhh,
                             const float* __restrict__ decWih, const float* __restrict__ decWhh,
                             const float* __restrict__ decbih, const float* __restrict__ decbhh,
                             __bf16* __restrict__ encW, __bf16* __restrict__ decW,
                             float* __restrict__ encB, float* __restrict__ decB)
{
    int idx = blockIdx.x * blockDim.x + threadIdx.x;
    if (idx < G4_ * KAUG) {
        int n = idx / KAUG, k = idx % KAUG;
        float we = (k < H_) ? encWhh[(size_t)n * H_ + k]
                            : ((k < H_ + IN_) ? encWih[(size_t)n * IN_ + (k - H_)] : 0.f);
        float wd = (k < H_) ? decWhh[(size_t)n * H_ + k]
                            : ((k == H_) ? decWih[n] : 0.f);   // OUT == 1
        encW[idx] = (__bf16)we;
        decW[idx] = (__bf16)wd;
    }
    if (idx < G4_) {
        encB[idx] = encbih[idx] + encbhh[idx];
        decB[idx] = decbih[idx] + decbhh[idx];
    }
}

// ---------------- one fused LSTM step (WMMA gates + in-register gate fusion) -------
// Each wave owns 32 hidden columns (2 N-tiles), all 4 gates of those columns.
// c frags persist in registers in C-matrix layout: lane&15 = N, vgpr r (+8*hi) = M.
__device__ __forceinline__ void lstm_step(const __bf16* __restrict__ Waug,
                                          __bf16* aA,                 // LDS [16][KAUG]
                                          const float (&bias)[2][4],
                                          v8f (&c)[2],
                                          int m, int hi, int wave)
{
    v8f acc[2][4];
#pragma unroll
    for (int it = 0; it < 2; ++it)
#pragma unroll
        for (int g = 0; g < 4; ++g) acc[it][g] = (v8f){0,0,0,0,0,0,0,0};

    const int abase = m * KAUG + hi * 8;
#pragma unroll 1
    for (int kt = 0; kt < KT_; ++kt) {
        const int k0 = kt * 32;
        // A fragment (16x32 bf16): lane m holds K = (hi*8)+{0..7,16..23} of row m
        v8bf alo = *(const v8bf*)(aA + abase + k0);
        v8bf ahi = *(const v8bf*)(aA + abase + k0 + 16);
        v16bf a;
#pragma unroll
        for (int j = 0; j < 8; ++j) { a[j] = alo[j]; a[j + 8] = ahi[j]; }
#pragma unroll
        for (int it = 0; it < 2; ++it) {
#pragma unroll
            for (int g = 0; g < 4; ++g) {
                // B fragment (32x16 bf16): lane holds N=lane&15, K = (hi*8)+{0..7,16..23}
                const int ncol = g * H_ + wave * 32 + it * 16 + m;
                const __bf16* bp = Waug + (size_t)ncol * KAUG + k0 + hi * 8;
                v8bf blo = *(const v8bf*)bp;
                v8bf bhi = *(const v8bf*)(bp + 16);
                v16bf b;
#pragma unroll
                for (int j = 0; j < 8; ++j) { b[j] = blo[j]; b[j + 8] = bhi[j]; }
                acc[it][g] = __builtin_amdgcn_wmma_f32_16x16x32_bf16(
                    false, a, false, b, (short)0, acc[it][g], false, false);
            }
        }
    }
    __syncthreads();   // all waves done reading old h before anyone overwrites it
#pragma unroll
    for (int it = 0; it < 2; ++it) {
        const int nloc = wave * 32 + it * 16 + m;   // hidden column (C-frag: lane = N)
#pragma unroll
        for (int r = 0; r < 8; ++r) {               // C-frag: vgpr r -> batch row r+8*hi
            float gi = sig_ (acc[it][0][r] + bias[it][0]);
            float gf = sig_ (acc[it][1][r] + bias[it][1]);
            float gg = tanh_(acc[it][2][r] + bias[it][2]);
            float go = sig_ (acc[it][3][r] + bias[it][3]);
            float cn = gf * c[it][r] + gi * gg;
            c[it][r] = cn;
            aA[(r + 8 * hi) * KAUG + nloc] = (__bf16)(go * tanh_(cn));
        }
    }
}

// ---------------- persistent encoder+decoder kernel: 1 block = 16 batch rows ------
__global__ __launch_bounds__(512)
void lstm_ed_main(const float* __restrict__ x,
                  const __bf16* __restrict__ encW, const float* __restrict__ encB,
                  const __bf16* __restrict__ decW, const float* __restrict__ decB,
                  const float* __restrict__ outW, const float* __restrict__ outb,
                  float* __restrict__ out)
{
    __shared__ __bf16 aA[16 * KAUG];     // h (cols 0..511) | x or y (cols 512..519) | pad
    __shared__ float  red[256];
    __shared__ float  predbuf[16];
    __shared__ float  outWs[H_];

    const int tid  = threadIdx.x;
    const int lane = tid & 31;
    const int wave = tid >> 5;
    const int m    = lane & 15;
    const int hi   = lane >> 4;
    const int b0   = blockIdx.x * 16;

    for (int i = tid; i < 16 * KAUG; i += 512) aA[i] = (__bf16)0.f;   // h0 = 0
    for (int i = tid; i < H_; i += 512) outWs[i] = outW[i];

    float bias[2][4];
    v8f   c[2];
#pragma unroll
    for (int it = 0; it < 2; ++it) {
        c[it] = (v8f){0,0,0,0,0,0,0,0};                               // c0 = 0
#pragma unroll
        for (int g = 0; g < 4; ++g)
            bias[it][g] = encB[g * H_ + wave * 32 + it * 16 + m];
    }
    __syncthreads();

    // ===== encoder: scan over S timesteps =====
#pragma unroll 1
    for (int t = 0; t < S_; ++t) {
        if (tid < 128) {                                              // stage x_t
            int mm = tid >> 3, j = tid & 7;
            aA[mm * KAUG + H_ + j] = (__bf16)x[((size_t)(b0 + mm) * S_ + t) * IN_ + j];
        }
        __syncthreads();
        lstm_step(encW, aA, bias, c, m, hi, wave);
    }

    // ===== decoder: autoregressive =====
#pragma unroll
    for (int it = 0; it < 2; ++it)
#pragma unroll
        for (int g = 0; g < 4; ++g)
            bias[it][g] = decB[g * H_ + wave * 32 + it * 16 + m];

    if (tid < 128) {                                                  // y0 = 0
        int mm = tid >> 3, j = tid & 7;
        aA[mm * KAUG + H_ + j] = (__bf16)0.f;
    }
    __syncthreads();

#pragma unroll 1
    for (int f = 0; f < F_; ++f) {
        lstm_step(decW, aA, bias, c, m, hi, wave);
        __syncthreads();                                              // h writes visible
        if (tid < 256) {                                              // pred = h @ outW^T + b
            int mm = tid >> 4, seg = tid & 15;
            const v8bf* hrow = (const v8bf*)(aA + mm * KAUG + seg * 32);
            float p = 0.f;
#pragma unroll
            for (int v = 0; v < 4; ++v) {
                v8bf hv = hrow[v];
#pragma unroll
                for (int k = 0; k < 8; ++k)
                    p += (float)hv[k] * outWs[seg * 32 + v * 8 + k];
            }
            red[tid] = p;
        }
        __syncthreads();
        if (tid < 16) {
            float s = 0.f;
#pragma unroll
            for (int q = 0; q < 16; ++q) s += red[tid * 16 + q];
            s += outb[0];
            out[(size_t)(b0 + tid) * F_ + f] = s;
            predbuf[tid] = s;
        }
        __syncthreads();
        if (tid < 128) {                                              // feed y back
            int mm = tid >> 3, j = tid & 7;
            aA[mm * KAUG + H_ + j] = (j == 0) ? (__bf16)predbuf[mm] : (__bf16)0.f;
        }
        __syncthreads();
    }
}

// ---------------- launch ----------------
extern "C" void kernel_launch(void* const* d_in, const int* in_sizes, int n_in,
                              void* d_out, int out_size, void* d_ws, size_t ws_size,
                              hipStream_t stream)
{
    const float* x      = (const float*)d_in[0];
    const float* encWih = (const float*)d_in[1];
    const float* encWhh = (const float*)d_in[2];
    const float* encbih = (const float*)d_in[3];
    const float* encbhh = (const float*)d_in[4];
    const float* decWih = (const float*)d_in[5];
    const float* decWhh = (const float*)d_in[6];
    const float* decbih = (const float*)d_in[7];
    const float* decbhh = (const float*)d_in[8];
    const float* outW   = (const float*)d_in[9];
    const float* outb   = (const float*)d_in[10];

    // workspace layout: encW_aug | decW_aug (bf16) | encB | decB (f32)  (~4.3 MB)
    __bf16* encW = (__bf16*)d_ws;
    __bf16* decW = encW + (size_t)G4_ * KAUG;
    float*  encB = (float*)(decW + (size_t)G4_ * KAUG);
    float*  decB = encB + G4_;

    const int total = G4_ * KAUG;
    pack_weights<<<(total + 255) / 256, 256, 0, stream>>>(
        encWih, encWhh, encbih, encbhh, decWih, decWhh, decbih, decbhh,
        encW, decW, encB, decB);

    lstm_ed_main<<<B_ / 16, 512, 0, stream>>>(
        x, encW, encB, decW, decB, outW, outb, (float*)d_out);
}
